// MultiHeadAttention_35416300323297
// MI455X (gfx1250) — compile-verified, matrix operand order checked by reading, not run on hardware
//
#include <hip/hip_runtime.h>
#include <stdint.h>

// ---------------- problem constants ----------------
#define DMODEL 1024
#define NHEAD 16
#define HD 64
#define SEQ 2048
#define BATCH 2
#define MTOT (BATCH * SEQ) // 4096
#define SCALE 0.125f       // 1/sqrt(64)

typedef _Float16 half_t;
typedef _Float16 v16h __attribute__((ext_vector_type(16)));
typedef float v8f __attribute__((ext_vector_type(8)));
typedef unsigned int u32x4 __attribute__((ext_vector_type(4)));
typedef int i32x4 __attribute__((ext_vector_type(4)));
typedef int i32x8 __attribute__((ext_vector_type(8)));

union FragAB { v16h v; uint32_t u[8]; };

__device__ __forceinline__ v8f zero8() {
  v8f z = {0.f, 0.f, 0.f, 0.f, 0.f, 0.f, 0.f, 0.f};
  return z;
}
__device__ __forceinline__ v8f wmma_f16(const FragAB& a, const FragAB& b, v8f c) {
  return __builtin_amdgcn_wmma_f32_16x16x32_f16(false, a.v, false, b.v,
                                                (short)0, c, false, false);
}
__device__ __forceinline__ uint16_t h2u(half_t h) {
  union { half_t h; uint16_t u; } c; c.h = h; return c.u;
}
__device__ __forceinline__ half_t u2h(uint16_t u) {
  union { half_t h; uint16_t u; } c; c.u = u; return c.h;
}
__device__ __forceinline__ uint32_t pack2(float a, float b) {
  return (uint32_t)h2u((half_t)a) | ((uint32_t)h2u((half_t)b) << 16);
}

// ------------------------------------------------------------
// Tensor Data Mover: 2-D tile (f16 elements) global -> LDS with
// row padding (reproduces our +8-half LDS stride in hardware).
// D# layout per CDNA5 ISA 8.3/8.4. Groups 2/3 zero (2-D tensor).
// ------------------------------------------------------------
__device__ __forceinline__ void tdm_load_2d_f16(
    uint32_t ldsOff, const half_t* gptr,
    uint32_t tensorDim0, uint32_t tensorDim1, uint32_t tile0, uint32_t tile1,
    uint32_t stride0, uint32_t padInterval, uint32_t padAmount) {
  uint64_t ga = (uint64_t)(uintptr_t)gptr;
  u32x4 g0;
  g0[0] = 1u; // count=1, is_restore=0, gather off
  g0[1] = ldsOff;
  g0[2] = (uint32_t)(ga & 0xFFFFFFFFu);
  g0[3] = (uint32_t)((ga >> 32) & 0x01FFFFFFu) | (2u << 30); // type=2 (image)
  i32x8 g1;
  g1[0] = (int)((1u << 16) |            // data_size = 1 -> 2 bytes
                (1u << 20) |            // pad_enable
                (padInterval << 22) | (padAmount << 25));
  g1[1] = (int)((tensorDim0 & 0xFFFFu) << 16);
  g1[2] = (int)(((tensorDim0 >> 16) & 0xFFFFu) | ((tensorDim1 & 0xFFFFu) << 16));
  g1[3] = (int)(((tensorDim1 >> 16) & 0xFFFFu) | ((tile0 & 0xFFFFu) << 16));
  g1[4] = (int)(tile1 & 0xFFFFu); // tile_dim2 = 0
  g1[5] = (int)stride0;           // tensor_dim0_stride (48b, hi = 0)
  g1[6] = 0;
  g1[7] = 0;
  i32x4 z4 = {0, 0, 0, 0};
#if __clang_major__ >= 23
  i32x8 z8 = {0, 0, 0, 0, 0, 0, 0, 0};
  __builtin_amdgcn_tensor_load_to_lds(g0, g1, z4, z4, z8, 0);
#else
  __builtin_amdgcn_tensor_load_to_lds(g0, g1, z4, z4, 0);
#endif
}

// ============================================================
// GEMM: out[M,N] = A[M,K] @ W[N,K]^T + bias   (K = N = 1024)
// AHALF=0: A = f32, out -> f16 ws in [B,H,S,HD] head layout (MODE<3)
// AHALF=1: A = f16 attn-out, out -> f32 d_out (MODE=3)
// Tile 128x128x32, 256 threads = 8 waves (2 x 4), wave tile 64x32.
// Double-buffered: next tile's global loads fly under current WMMAs.
// ============================================================
#define BM 128
#define BN 128
#define BK 32
#define LDT 20 // uint stride per row (= 40 halves: +8 pad, conflict-free)

template <int AHALF, int MODE>
__global__ void __launch_bounds__(256)
gemm_xwT(const float* __restrict__ Xf, const half_t* __restrict__ Xh,
         const float* __restrict__ Wm, const float* __restrict__ bias,
         half_t* __restrict__ outH, float* __restrict__ outF) {
  __shared__ uint32_t As[BM * LDT];
  __shared__ uint32_t Bs[BN * LDT];
  const int tid = threadIdx.x;
  const int lane = tid & 31, w = tid >> 5;
  const int m0 = blockIdx.y * BM, n0 = blockIdx.x * BN;
  const int wm = (w >> 2) * 64;
  const int wn = (w & 3) * 32;
  const int lm = lane & 15;
  const int hi4 = (lane >> 4) * 4;
  const int hiRow = (lane >> 4) * 8;

  v8f acc[4][2];
#pragma unroll
  for (int i = 0; i < 4; ++i)
#pragma unroll
    for (int j = 0; j < 2; ++j) acc[i][j] = zero8();

  // register staging for the incoming tile (all loads issued back-to-back)
  float2 aR[8], bR[8];
  uint32_t aU[8];

  auto fetch = [&](int k0) {
#pragma unroll
    for (int it = 0; it < 8; ++it) {
      int p = tid + it * 256;
      int row = p >> 4;
      int kp = p & 15;
      if (AHALF)
        aU[it] = *(const uint32_t*)(Xh + (size_t)(m0 + row) * DMODEL + k0 + kp * 2);
      else
        aR[it] = *(const float2*)(Xf + (size_t)(m0 + row) * DMODEL + k0 + kp * 2);
      bR[it] = *(const float2*)(Wm + (size_t)(n0 + row) * DMODEL + k0 + kp * 2);
    }
  };
  auto stage = [&]() {
#pragma unroll
    for (int it = 0; it < 8; ++it) {
      int p = tid + it * 256;
      int row = p >> 4;
      int kp = p & 15;
      As[row * LDT + kp] = AHALF ? aU[it] : pack2(aR[it].x, aR[it].y);
      Bs[row * LDT + kp] = pack2(bR[it].x, bR[it].y);
    }
  };

  fetch(0);
  for (int k0 = 0; k0 < DMODEL; k0 += BK) {
    __syncthreads(); // previous iteration done reading LDS
    stage();
    __syncthreads();
    if (k0 + BK < DMODEL) fetch(k0 + BK); // loads in flight under the WMMAs

    FragAB bfr[2];
#pragma unroll
    for (int j = 0; j < 2; ++j) {
      int n = wn + j * 16 + lm;
#pragma unroll
      for (int q = 0; q < 8; ++q) {
        int kp = q + ((q >> 2) << 2) + hi4;
        bfr[j].u[q] = Bs[n * LDT + kp];
      }
    }
#pragma unroll
    for (int i = 0; i < 4; ++i) {
      FragAB afr;
      int m = wm + i * 16 + lm;
#pragma unroll
      for (int q = 0; q < 8; ++q) {
        int kp = q + ((q >> 2) << 2) + hi4;
        afr.u[q] = As[m * LDT + kp];
      }
#pragma unroll
      for (int j = 0; j < 2; ++j) acc[i][j] = wmma_f16(afr, bfr[j], acc[i][j]);
    }
  }

#pragma unroll
  for (int i = 0; i < 4; ++i)
#pragma unroll
    for (int j = 0; j < 2; ++j) {
      int col = n0 + wn + j * 16 + lm;
      float bv = bias[col];
#pragma unroll
      for (int r = 0; r < 8; ++r) {
        int gi = m0 + wm + i * 16 + r + hiRow;
        float val = acc[i][j][r] + bv;
        if (MODE == 3) {
          outF[(size_t)gi * DMODEL + col] = val;
        } else {
          int b = gi >> 11, s = gi & (SEQ - 1);
          int h = col >> 6, hd = col & 63;
          outH[(((size_t)b * NHEAD + h) * SEQ + s) * HD + hd] = (half_t)val;
        }
      }
    }
}

// ============================================================
// Attention: one block per (b, h, 32-row query tile).
// 32 x 2048 f16 score strip lives in LDS (CDNA5 320KB LDS).
// Q/K tiles arrive via the Tensor Data Mover (hardware-padded rows);
// softmax in f32; probs streamed NT to d_out; f16 probs -> P@V WMMA.
// ============================================================
#define SSTR 1028 // uint stride of a score row = 2056 halves (+8 pad)
#define QSTR 36   // uint stride = 72 halves (+8 pad)
#define VSTR 20   // uint stride = 40 halves (+8 pad)
// 64 f16 per row = 128 B = 16 x 8B units -> pad_interval = 4 (1<<4 = 16)
// pad 8 halves = 4 DWORDs -> pad_amount code 3
#define PAD_I 4u
#define PAD_A 3u

__global__ void __launch_bounds__(256)
attn(const half_t* __restrict__ qh, const half_t* __restrict__ kh,
     const half_t* __restrict__ vh, half_t* __restrict__ oh,
     float* __restrict__ Pout) {
  __shared__ uint32_t Sc[32 * SSTR]; // 131584 B
  __shared__ uint32_t Qs[32 * QSTR]; //   4608 B
  __shared__ uint32_t Ks[64 * QSTR]; //   9216 B
  __shared__ uint32_t Vt[64 * VSTR]; //   5120 B
  __shared__ float red[32][9];

  const int tid = threadIdx.x, lane = tid & 31, w = tid >> 5;
  const int i0 = blockIdx.x * 32;
  const int h = blockIdx.y, b = blockIdx.z;
  const size_t headOff = ((size_t)b * NHEAD + h) * SEQ * HD;
  const half_t* Q = qh + headOff;
  const half_t* K = kh + headOff;
  const half_t* V = vh + headOff;
  int jCeil = ((i0 + 32 + 63) >> 6) << 6;
  if (jCeil > SEQ) jCeil = SEQ;

  const int lm = lane & 15;
  const int hi4 = (lane >> 4) * 4;
  const int hiRow = (lane >> 4) * 8;
  uint16_t* sc16 = (uint16_t*)Sc;
  const uint32_t qsOff = (uint32_t)(uintptr_t)(void*)&Qs[0];
  const uint32_t ksOff = (uint32_t)(uintptr_t)(void*)&Ks[0];

  // ---- Q tile 32x64 via TDM (padded rows) ----
  if (w == 0) {
    tdm_load_2d_f16(qsOff, Q + (size_t)i0 * HD, HD, SEQ - i0, HD, 32, HD,
                    PAD_I, PAD_A);
  }

  // ---- phase 1: scores = scale * Q K^T with causal mask, into LDS f16 ----
  const int jfr = w & 3, rfr = w >> 2;
  for (int jt = 0; jt < jCeil; jt += 64) {
    __syncthreads(); // everyone done reading the previous K tile
    if (w == 0) {
      tdm_load_2d_f16(ksOff, K + (size_t)jt * HD, HD, SEQ - jt, HD, 64, HD,
                      PAD_I, PAD_A);
      __builtin_amdgcn_s_wait_tensorcnt(0); // covers Q load too (in-order)
    }
    __syncthreads(); // TDM data visible to all waves

    v8f sacc = zero8();
#pragma unroll
    for (int kk = 0; kk < 2; ++kk) { // HD = 64 = 2 x K32
      FragAB afr, bfr;
#pragma unroll
      for (int q = 0; q < 8; ++q) {
        int kp = q + ((q >> 2) << 2) + hi4 + kk * 16;
        afr.u[q] = Qs[(rfr * 16 + lm) * QSTR + kp];
        bfr.u[q] = Ks[(jfr * 16 + lm) * QSTR + kp];
      }
      sacc = wmma_f16(afr, bfr, sacc);
    }
#pragma unroll
    for (int r = 0; r < 8; ++r) {
      int m = rfr * 16 + r + hiRow;
      int jg = jt + jfr * 16 + lm;
      int gi = i0 + m;
      float v = sacc[r] * SCALE;
      half_t hv = (jg > gi) ? (half_t)(-__builtin_inff()) : (half_t)v;
      sc16[m * (SSTR * 2) + jg] = h2u(hv);
    }
  }
  __syncthreads();

  // ---- phase 2: softmax over each row (8 threads per row) ----
  {
    int r = tid >> 3, sub = tid & 7;
    float mx = -__builtin_inff();
    for (int j = sub; j < jCeil; j += 8) {
      float v = (float)u2h(sc16[r * (SSTR * 2) + j]);
      mx = fmaxf(mx, v);
    }
    red[r][sub] = mx;
    __syncthreads();
    float rmx = -__builtin_inff();
#pragma unroll
    for (int q = 0; q < 8; ++q) rmx = fmaxf(rmx, red[r][q]);
    __syncthreads();
    float sum = 0.f;
    for (int j = sub; j < jCeil; j += 8) {
      float v = (float)u2h(sc16[r * (SSTR * 2) + j]);
      float e = __expf(v - rmx); // exp(-inf)=0 handles the causal mask
      sc16[r * (SSTR * 2) + j] = h2u((half_t)e);
      sum += e;
    }
    red[r][sub] = sum;
    __syncthreads();
    float tot = 0.f;
#pragma unroll
    for (int q = 0; q < 8; ++q) tot += red[r][q];
    if (sub == 0) red[r][8] = 1.0f / tot;
  }
  __syncthreads();

  // ---- phase 3: normalize; stream f32 probs to global (NT), f16 to LDS ----
  {
    float* Prow = Pout + (((size_t)b * NHEAD + h) * SEQ + i0) * SEQ;
    for (int r = 0; r < 32; ++r) {
      float inv = red[r][8];
      for (int j = tid; j < SEQ; j += 256) { // coalesced per row
        float p = 0.f;
        if (j < jCeil) {
          float e = (float)u2h(sc16[r * (SSTR * 2) + j]);
          p = e * inv;
          sc16[r * (SSTR * 2) + j] = h2u((half_t)p);
        }
        __builtin_nontemporal_store(p, Prow + (size_t)r * SEQ + j);
      }
    }
  }
  __syncthreads();

  // ---- phase 4: O = P @ V  (one 16x16 output fragment per wave) ----
  const int dfr = w & 3, rfr2 = w >> 2;
  v8f oacc = zero8();
  uint16_t* vt16 = (uint16_t*)Vt;
  for (int jt = 0; jt < jCeil; jt += 32) {
    __syncthreads();
#pragma unroll
    for (int it = 0; it < 8; ++it) { // V tile 32x64 transposed into Vt[d][j]
      int e = tid + it * 256;
      int jl = e >> 6, d = e & 63;
      vt16[d * (VSTR * 2) + jl] = h2u(V[(size_t)(jt + jl) * HD + d]);
    }
    __syncthreads();
    FragAB afr, bfr;
#pragma unroll
    for (int q = 0; q < 8; ++q) {
      int kp = q + ((q >> 2) << 2) + hi4;
      afr.u[q] = Sc[(rfr2 * 16 + lm) * SSTR + (jt >> 1) + kp];
      bfr.u[q] = Vt[(dfr * 16 + lm) * VSTR + kp];
    }
    oacc = wmma_f16(afr, bfr, oacc);
  }
#pragma unroll
  for (int r = 0; r < 8; ++r) {
    int m = rfr2 * 16 + r + hiRow;
    int gi = i0 + m;
    int dcol = dfr * 16 + lm;
    oh[((size_t)b * SEQ + gi) * DMODEL + h * HD + dcol] = (half_t)oacc[r];
  }
}

// ============================================================
extern "C" void kernel_launch(void* const* d_in, const int* in_sizes, int n_in,
                              void* d_out, int out_size, void* d_ws, size_t ws_size,
                              hipStream_t stream) {
  const float* query = (const float*)d_in[0];
  const float* Wq = (const float*)d_in[1];
  const float* bq = (const float*)d_in[2];
  const float* Wk = (const float*)d_in[3];
  const float* bk = (const float*)d_in[4];
  const float* Wv = (const float*)d_in[5];
  const float* bv = (const float*)d_in[6];
  const float* Wo = (const float*)d_in[7];
  const float* bo = (const float*)d_in[8];
  // d_in[9] = attn_mask (bool causal) — causality applied analytically.

  float* out = (float*)d_out;                       // [B,S,D] f32
  float* Pout = out + (size_t)BATCH * SEQ * DMODEL; // [B,H,S,S] f32 weights

  const size_t bufElems = (size_t)BATCH * SEQ * DMODEL; // 4,194,304 halves
  half_t* qh = (half_t*)d_ws;
  half_t* kh = qh + bufElems;
  half_t* vh = kh + bufElems;
  half_t* oh = vh + bufElems;

  dim3 gG(DMODEL / BN, MTOT / BM, 1); // (8, 32)
  gemm_xwT<0, 0><<<gG, 256, 0, stream>>>(query, nullptr, Wq, bq, qh, nullptr);
  gemm_xwT<0, 1><<<gG, 256, 0, stream>>>(query, nullptr, Wk, bk, kh, nullptr);
  gemm_xwT<0, 2><<<gG, 256, 0, stream>>>(query, nullptr, Wv, bv, vh, nullptr);

  attn<<<dim3(SEQ / 32, NHEAD, BATCH), 256, 0, stream>>>(qh, kh, vh, oh, Pout);

  gemm_xwT<1, 3><<<gG, 256, 0, stream>>>(nullptr, oh, Wo, bo, nullptr, out);
}
